// GraphConvLayer_51694226375550
// MI455X (gfx1250) — matrix operand order repaired and checked by reading, not compiled, not run on hardware
//
#include <hip/hip_runtime.h>

typedef __attribute__((ext_vector_type(16))) __bf16 v16bf;
typedef __attribute__((ext_vector_type(4)))  __bf16 v4bf;
typedef __attribute__((ext_vector_type(8)))  float  v8f;

#define NROWS   8192
#define NCOLS   128
#define KC      64
#define TSTRIDE 72   // bf16 per LDS row: 144B = 9*16B -> 16B-aligned b128 frag loads

// out[m, 0:128] = (A[m, 0:K] @ B[0:K, 0:128]) * (NORM ? 1/rowsum(A[m,:]) : 1)
// A: rows x lda fp32 row-major.  B: K x 128 fp32 row-major (L2-resident).
// One wave per 16x128 output tile; 3-term bf16-split WMMA, fp32 accumulate.
template<bool NORM>
__global__ __launch_bounds__(256)
void gcn_gemm_kernel(const float* __restrict__ A, int lda, int K,
                     const float* __restrict__ B,
                     float* __restrict__ out)
{
    __shared__ __bf16 Thi[NCOLS * TSTRIDE];
    __shared__ __bf16 Tlo[NCOLS * TSTRIDE];

    const int tid  = threadIdx.x;
    const int wave = tid >> 5;
    const int lane = tid & 31;
    const int hf   = lane >> 4;      // lane half (0/1)
    const int l16  = lane & 15;

    const int m0 = blockIdx.x * 128 + wave * 16;     // 16 rows per wave

    // staging assignment: each thread owns one column n, 4 consecutive k per step
    const int n_s  = tid & 127;
    const int k0_s = (tid >> 7) * 4;

    v8f acc[8];
    #pragma unroll
    for (int i = 0; i < 8; ++i) acc[i] = (v8f){};
    float rowAcc = 0.0f;

    const float* Arow = A + (size_t)(m0 + l16) * (size_t)lda;

    for (int kc = 0; kc < K; kc += KC) {
        if (kc + KC < K)  // hint next A chunk toward the caches
            __builtin_prefetch(Arow + kc + KC + hf * 8, 0, 1);

        // ---- stage B[kc:kc+64, 0:128] transposed into LDS as bf16 hi/lo ----
        #pragma unroll
        for (int i = 0; i < 8; ++i) {
            const int k = k0_s + i * 8;
            float f[4];
            #pragma unroll
            for (int j = 0; j < 4; ++j)                       // coalesced 128B/row
                f[j] = B[(size_t)(kc + k + j) * NCOLS + n_s];
            v4bf h, l;
            #pragma unroll
            for (int j = 0; j < 4; ++j) {
                __bf16 hj = (__bf16)f[j];
                h[j] = hj;
                l[j] = (__bf16)(f[j] - (float)hj);
            }
            *(v4bf*)&Thi[n_s * TSTRIDE + k] = h;              // 8B ds stores
            *(v4bf*)&Tlo[n_s * TSTRIDE + k] = l;
        }
        __syncthreads();

        // ---- compute: 2 k-steps of 32, 8 n-tiles, 3 split terms each ----
        #pragma unroll
        for (int ks = 0; ks < KC; ks += 32) {
            // A fragment (16-bit A 16x32, contiguous k): lane L, elem e ->
            //   k = e + 16*(L>=16)
            v16bf ahi, alo;
            {
                const float* p = Arow + kc + ks + hf * 16;
                const float4 r0 = *(const float4*)(p + 0);
                const float4 r1 = *(const float4*)(p + 4);
                const float4 r2 = *(const float4*)(p + 8);
                const float4 r3 = *(const float4*)(p + 12);
                float f[16] = { r0.x,r0.y,r0.z,r0.w, r1.x,r1.y,r1.z,r1.w,
                                r2.x,r2.y,r2.z,r2.w, r3.x,r3.y,r3.z,r3.w };
                #pragma unroll
                for (int e = 0; e < 16; ++e) {
                    if (NORM) rowAcc += f[e];                 // fused degree
                    __bf16 he = (__bf16)f[e];
                    ahi[e] = he;
                    alo[e] = (__bf16)(f[e] - (float)he);
                }
            }
            #pragma unroll
            for (int nt = 0; nt < 8; ++nt) {
                // B fragment (32x16): lane L, elem e -> k = e + 16*(L>=16), n = nt*16 + L%16
                const int boff = (nt * 16 + l16) * TSTRIDE + ks + hf * 16;
                v16bf bhi = *(const v16bf*)&Thi[boff];        // 2x aligned ds_load_b128
                v16bf blo = *(const v16bf*)&Tlo[boff];
                acc[nt] = __builtin_amdgcn_wmma_f32_16x16x32_bf16(
                              false, ahi, false, bhi, (short)0, acc[nt], false, false);
                acc[nt] = __builtin_amdgcn_wmma_f32_16x16x32_bf16(
                              false, ahi, false, blo, (short)0, acc[nt], false, false);
                acc[nt] = __builtin_amdgcn_wmma_f32_16x16x32_bf16(
                              false, alo, false, bhi, (short)0, acc[nt], false, false);
            }
        }
        __syncthreads();
    }

    // ---- row normalization: full rowsum = this lane's partial + partner lane ----
    float inv[8];
    if (NORM) {
        const float deg = rowAcc + __shfl_xor(rowAcc, 16, 32);  // lane L: deg of row L%16
        #pragma unroll
        for (int r = 0; r < 8; ++r) {
            const float d0 = __shfl(deg, r,     32);            // rows 0..7
            const float d1 = __shfl(deg, r + 8, 32);            // rows 8..15
            inv[r] = 1.0f / (hf ? d1 : d0);
        }
    } else {
        #pragma unroll
        for (int r = 0; r < 8; ++r) inv[r] = 1.0f;
    }

    // ---- store: C/D layout VGPR r, lane L -> row = r + 8*(L>=16), col = nt*16 + L%16
    #pragma unroll
    for (int nt = 0; nt < 8; ++nt) {
        #pragma unroll
        for (int r = 0; r < 8; ++r) {
            const int row = m0 + r + 8 * hf;
            const int col = nt * 16 + l16;
            out[(size_t)row * NCOLS + col] = acc[nt][r] * inv[r];
        }
    }
}

extern "C" void kernel_launch(void* const* d_in, const int* in_sizes, int n_in,
                              void* d_out, int out_size, void* d_ws, size_t ws_size,
                              hipStream_t stream) {
    const float* x   = (const float*)d_in[0];  // [8192, 128]
    const float* adj = (const float*)d_in[1];  // [8192, 8192]
    const float* w   = (const float*)d_in[2];  // [128, 128]
    float*       out = (float*)d_out;          // [8192, 128]
    float*       xw  = (float*)d_ws;           // scratch: [8192, 128] fp32 = 4 MB

    dim3 grid(NROWS / 128), block(256);
    // Pass 1: xw = x @ W          (tiny; fills L2-resident operand for pass 2)
    gcn_gemm_kernel<false><<<grid, block, 0, stream>>>(x, NCOLS, NCOLS, w, xw);
    // Pass 2: out = (A @ xw) / rowsum(A)   (single streaming pass over 256 MB A)
    gcn_gemm_kernel<true ><<<grid, block, 0, stream>>>(adj, NROWS, NROWS, xw, out);
}
